// SimplifiedNet_69621419868686
// MI455X (gfx1250) — matrix operand order repaired and checked by reading, not compiled, not run on hardware
//
#include <hip/hip_runtime.h>
#include <hip/hip_bf16.h>
#include <stdint.h>

typedef __attribute__((ext_vector_type(16))) __bf16 v16bf;
typedef __attribute__((ext_vector_type(8)))  __bf16 v8bf;
typedef __attribute__((ext_vector_type(8)))  float  v8f;

// CDNA5 async global->LDS path (gfx1250): guarded so we fall back to
// synchronous LDS staging if the toolchain lacks the builtins.
#if defined(__HIP_DEVICE_COMPILE__) && \
    __has_builtin(__builtin_amdgcn_global_load_async_to_lds_b128) && \
    __has_builtin(__builtin_amdgcn_s_wait_asynccnt)
#define ATHENA_ASYNC_LDS 1
#else
#define ATHENA_ASYNC_LDS 0
#endif

#if ATHENA_ASYNC_LDS
// Probe-derived prototype: (int4 addrspace(1)*, int4 addrspace(3)*, imm, imm)
typedef int v4i __attribute__((vector_size(16)));
typedef __attribute__((address_space(1))) v4i as1_v4i;
typedef __attribute__((address_space(3))) v4i as3_v4i;
#endif

namespace cfg {
constexpr int S = 128, B = 64, V = 16000, E = 512, H = 1024, L = 128;
constexpr int IO = 2 * H;     // 2048
constexpr int G  = 4 * H;     // 4096 gate width
constexpr int KC1 = E + H;    // 1536  encoder layer-1 A width  [e_t | h1]
constexpr int KC2 = H + H;    // 2048  encoder layer-2 A width  [h1_n | h2]
constexpr int KCD = E + H;    // 1536  decoder A width          [e_t | hd]
}

__device__ __forceinline__ float sigf(float x) { return 1.0f / (1.0f + __expf(-x)); }

// One 16x16x32 bf16 WMMA step: A fragment from global, B fragment from LDS tile.
__device__ __forceinline__ void mma_step(v8f& acc, const __bf16* __restrict__ ap, int k0,
                                         int khalf, const __bf16* __restrict__ bs,
                                         int lrow) {
  // A 16x32 fragment: lane (lrow,khalf) holds K = khalf*8..+7 and 16+khalf*8..+7
  v8bf lo = *reinterpret_cast<const v8bf*>(ap + k0 + khalf * 8);
  v8bf hi = *reinterpret_cast<const v8bf*>(ap + k0 + 16 + khalf * 8);
  v16bf a = __builtin_shufflevector(lo, hi, 0,1,2,3,4,5,6,7,8,9,10,11,12,13,14,15);
  // B 32x16 fragment: lane n = lrow holds Wtile[n][khalf*8..+7] and [16+khalf*8..+7]
  v8bf blo = *reinterpret_cast<const v8bf*>(bs + lrow * 32 + khalf * 8);
  v8bf bhi = *reinterpret_cast<const v8bf*>(bs + lrow * 32 + 16 + khalf * 8);
  v16bf b = __builtin_shufflevector(blo, bhi, 0,1,2,3,4,5,6,7,8,9,10,11,12,13,14,15);
  acc = __builtin_amdgcn_wmma_f32_16x16x32_bf16(false, a, false, b, (short)0, acc,
                                                false, false);
}

// ---------------- weight prep ----------------

__global__ void k_cvt_bf16(const float* __restrict__ in, __bf16* __restrict__ out, int n) {
  int i = blockIdx.x * blockDim.x + threadIdx.x;
  if (i < n) out[i] = (__bf16)in[i];
}

// out[N, Ka+Kb] = bf16([Wa | Wb]) row-wise concat (both row-major [N,K*])
__global__ void k_pack2(const float* __restrict__ Wa, const float* __restrict__ Wb,
                        __bf16* __restrict__ out, int N, int Ka, int Kb) {
  const int K = Ka + Kb;
  long i = (long)blockIdx.x * blockDim.x + threadIdx.x;
  if (i >= (long)N * K) return;
  int n = (int)(i / K), k = (int)(i % K);
  float v = (k < Ka) ? Wa[(long)n * Ka + k] : Wb[(long)n * Kb + (k - Ka)];
  out[i] = (__bf16)v;
}

__global__ void k_badd(const float* __restrict__ a, const float* __restrict__ b,
                       float* __restrict__ o, int n) {
  int i = blockIdx.x * blockDim.x + threadIdx.x;
  if (i < n) o[i] = a[i] + b[i];
}

// ---------------- embeddings ----------------

__global__ void k_embed_enc(const int* __restrict__ x, const float* __restrict__ emb,
                            __bf16* __restrict__ X) {
  using namespace cfg;
  long i = (long)blockIdx.x * blockDim.x + threadIdx.x;
  if (i >= (long)S * B * E) return;
  int sb = (int)(i / E), e = (int)(i % E);
  int tok = x[sb];
  X[i] = (__bf16)emb[(long)tok * E + e];
}

__global__ void k_embed_dec(const int* __restrict__ x, const float* __restrict__ emb,
                            __bf16* __restrict__ X) {
  using namespace cfg;
  long i = (long)blockIdx.x * blockDim.x + threadIdx.x;
  if (i >= (long)S * B * E) return;
  int sb = (int)(i / E), e = (int)(i % E);
  int tok = (sb < B) ? 1 : x[sb];   // dec_tokens = [BOS; x[1:]]
  X[i] = (__bf16)emb[(long)tok * E + e];
}

// ---------------- state init ----------------

__global__ void k_init(const float* __restrict__ h0, const float* __restrict__ c0,
                       const float* __restrict__ h02, const float* __restrict__ c02,
                       float* __restrict__ h1, float* __restrict__ c1,
                       float* __restrict__ h2, float* __restrict__ c2,
                       __bf16* __restrict__ A1, __bf16* __restrict__ A2,
                       const __bf16* __restrict__ Xenc) {
  using namespace cfg;
  int i = blockIdx.x * blockDim.x + threadIdx.x;
  if (i >= B * H) return;
  int b = i / H, j = i % H;
  h1[i] = h0[j]; c1[i] = c0[j]; h2[i] = h02[j]; c2[i] = c02[j];
  A1[b * KC1 + E + j] = (__bf16)h0[j];
  A2[b * KC2 + H + j] = (__bf16)h02[j];
  if (j < E) A1[b * KC1 + j] = Xenc[(size_t)b * E + j];  // e_0
}

// ---------------- unified LDS-tiled WMMA GEMM ----------------
// C[M,N] = A[M,K]bf16 @ W[N,K]bf16^T + bias[N]  (optional relu, f32 and/or bf16 out)
// block = (waves*32) threads; block tile M = 16*waves, N = 16.
// B tile (16x32 bf16, 1 KB) staged in LDS, shared by all waves; on toolchains with
// the gfx1250 async builtins the staging is double-buffered via ASYNCcnt.

__global__ __launch_bounds__(256)
void k_gemm(const __bf16* __restrict__ A, const __bf16* __restrict__ W,
            const float* __restrict__ bias, float* __restrict__ Cf,
            __bf16* __restrict__ Cb, int M, int N, int K, int relu) {
  const int tid   = threadIdx.x;
  const int wave  = tid >> 5;
  const int lane  = tid & 31;
  const int lrow  = lane & 15;
  const int khalf = lane >> 4;
  const int wpb   = blockDim.x >> 5;
  const int n0 = blockIdx.x * 16;
  const int m0 = (blockIdx.y * wpb + wave) * 16;

#if ATHENA_ASYNC_LDS
  __shared__ alignas(32) __bf16 Bs[2][16 * 32];
#else
  __shared__ alignas(32) __bf16 Bs[1][16 * 32];
#endif

  v8f acc;
  const float bv = bias[n0 + lrow];
#pragma unroll
  for (int r = 0; r < 8; ++r) acc[r] = bv;

  const __bf16* ap = A + (size_t)(m0 + lrow) * K;
  // staging threads (tid < 64): each copies 16 bytes of W[n0..n0+15][k0..k0+31]
  const int srow = tid >> 2, sseg = tid & 3;
  const __bf16* wrow = W + (size_t)(n0 + srow) * K + sseg * 8;

#if ATHENA_ASYNC_LDS
  const int nsteps = K >> 5;
  if (tid < 64) {
    __builtin_amdgcn_global_load_async_to_lds_b128(
        (as1_v4i*)(uintptr_t)wrow,
        (as3_v4i*)(uintptr_t)(unsigned)(uintptr_t)&Bs[0][srow * 32 + sseg * 8],
        0, 0);
  }
  for (int s = 0; s < nsteps; ++s) {
    const int buf = s & 1;
    __syncthreads();  // all readers done with buf^1 -> safe to overwrite it
    if (tid < 64) {
      if (s + 1 < nsteps) {
        __builtin_amdgcn_global_load_async_to_lds_b128(
            (as1_v4i*)(uintptr_t)(wrow + ((s + 1) << 5)),
            (as3_v4i*)(uintptr_t)(unsigned)(uintptr_t)
                &Bs[buf ^ 1][srow * 32 + sseg * 8],
            0, 0);
        __builtin_amdgcn_s_wait_asynccnt(1);  // current buf landed, next in flight
      } else {
        __builtin_amdgcn_s_wait_asynccnt(0);
      }
    }
    __syncthreads();  // current buf visible to all waves
    __builtin_prefetch(ap + (s << 5) + 128, 0, 0);  // A stream, 4 k-steps ahead
    mma_step(acc, ap, s << 5, khalf, &Bs[buf][0], lrow);
  }
#else
  for (int k0 = 0; k0 < K; k0 += 32) {
    __syncthreads();
    if (tid < 64) {
      *reinterpret_cast<uint4*>(&Bs[0][srow * 32 + sseg * 8]) =
          *reinterpret_cast<const uint4*>(wrow + k0);
    }
    __builtin_prefetch(ap + k0 + 128, 0, 0);
    __syncthreads();
    mma_step(acc, ap, k0, khalf, &Bs[0][0], lrow);
  }
#endif

  const int col = n0 + lrow;
#pragma unroll
  for (int r = 0; r < 8; ++r) {
    const int row = m0 + khalf * 8 + r;
    float v = acc[r];
    if (relu) v = v > 0.0f ? v : 0.0f;
    if (Cf) Cf[(size_t)row * N + col] = v;
    if (Cb) Cb[(size_t)row * N + col] = (__bf16)v;
  }
}

// ---------------- LSTM elementwise updates ----------------

__global__ void k_elem1(const float* __restrict__ g, const float* __restrict__ hp,
                        const float* __restrict__ cp, float* __restrict__ hn_out,
                        float* __restrict__ cn_out, const int* __restrict__ x, int t,
                        __bf16* __restrict__ A2, __bf16* __restrict__ A1,
                        const __bf16* __restrict__ Xenc) {
  using namespace cfg;
  int i = blockIdx.x * blockDim.x + threadIdx.x;
  if (i >= B * H) return;
  int b = i / H, j = i % H;
  const float* gb = g + (size_t)b * G;
  float ig = sigf(gb[j]);
  float fg = sigf(gb[H + j]);
  float gg = tanhf(gb[2 * H + j]);
  float og = sigf(gb[3 * H + j]);
  float cn = fg * cp[i] + ig * gg;
  float hn = og * tanhf(cn);
  bool m = (x[t * B + b] == 0);
  float hm = m ? hp[i] : hn;
  float cm = m ? cp[i] : cn;
  hn_out[i] = hm; cn_out[i] = cm;
  A2[b * KC2 + j]     = (__bf16)hn;  // layer-2 input uses UNmasked h_n (reference)
  A1[b * KC1 + E + j] = (__bf16)hm;  // next step layer-1 recurrent input
  if (j < E && t + 1 < S) A1[b * KC1 + j] = Xenc[((size_t)(t + 1) * B + b) * E + j];
}

__global__ void k_elem2(const float* __restrict__ g, const float* __restrict__ h2p,
                        const float* __restrict__ c2p, float* __restrict__ h2n_out,
                        float* __restrict__ c2n_out, const float* __restrict__ h1p,
                        const float* __restrict__ c1p, const int* __restrict__ x, int t,
                        __bf16* __restrict__ A2) {
  using namespace cfg;
  int i = blockIdx.x * blockDim.x + threadIdx.x;
  if (i >= B * H) return;
  int b = i / H, j = i % H;
  const float* gb = g + (size_t)b * G;
  float ig = sigf(gb[j]);
  float fg = sigf(gb[H + j]);
  float gg = tanhf(gb[2 * H + j]);
  float og = sigf(gb[3 * H + j]);
  float cn = fg * c2p[i] + ig * gg;
  float hn = og * tanhf(cn);
  bool m = (x[t * B + b] == 0);
  float hm = m ? h1p[i] : hn;   // faithful to reference: masked with layer-1 pre-states
  float cm = m ? c1p[i] : cn;
  h2n_out[i] = hm; c2n_out[i] = cm;
  A2[b * KC2 + H + j] = (__bf16)hm;
}

__global__ void k_elemd(const float* __restrict__ g, float* __restrict__ hd,
                        float* __restrict__ cd, int t, __bf16* __restrict__ Ad,
                        const __bf16* __restrict__ Xdec, __bf16* __restrict__ hs) {
  using namespace cfg;
  int i = blockIdx.x * blockDim.x + threadIdx.x;
  if (i >= B * H) return;
  int b = i / H, j = i % H;
  const float* gb = g + (size_t)b * G;
  float ig = sigf(gb[j]);
  float fg = sigf(gb[H + j]);
  float gg = tanhf(gb[2 * H + j]);
  float og = sigf(gb[3 * H + j]);
  float cn = fg * cd[i] + ig * gg;
  float hn = og * tanhf(cn);
  hd[i] = hn; cd[i] = cn;
  Ad[b * KCD + E + j] = (__bf16)hn;
  hs[((size_t)t * B + b) * H + j] = (__bf16)hn;
  if (j < E && t + 1 < S) Ad[b * KCD + j] = Xdec[((size_t)(t + 1) * B + b) * E + j];
}

// ---------------- CVAE glue ----------------

__global__ void k_hc(const float* __restrict__ h2, const float* __restrict__ c2,
                     float* __restrict__ out_hc, __bf16* __restrict__ hcb) {
  using namespace cfg;
  int i = blockIdx.x * blockDim.x + threadIdx.x;
  if (i >= B * IO) return;
  int b = i / IO, j = i % IO;
  float v = (j < H) ? h2[b * H + j] : c2[b * H + (j - H)];
  out_hc[i] = v; hcb[i] = (__bf16)v;
}

__global__ void k_z(const float* __restrict__ mean, const float* __restrict__ stdv,
                    const float* __restrict__ eps, float* __restrict__ zout,
                    __bf16* __restrict__ zb) {
  using namespace cfg;
  int i = blockIdx.x * blockDim.x + threadIdx.x;
  if (i >= B * L) return;
  float v = mean[i] + eps[i] * stdv[i];
  zout[i] = v; zb[i] = (__bf16)v;
}

__global__ void k_split(const float* __restrict__ rec, float* __restrict__ hd,
                        float* __restrict__ cd, __bf16* __restrict__ Ad,
                        const __bf16* __restrict__ Xdec) {
  using namespace cfg;
  int i = blockIdx.x * blockDim.x + threadIdx.x;
  if (i >= B * H) return;
  int b = i / H, j = i % H;
  float hv = rec[b * IO + j];
  float cv = rec[b * IO + H + j];
  hd[i] = hv; cd[i] = cv;
  Ad[b * KCD + E + j] = (__bf16)hv;
  if (j < E) Ad[b * KCD + j] = Xdec[(size_t)b * E + j];  // e_0 (BOS embedding)
}

// ---------------- host ----------------

extern "C" void kernel_launch(void* const* d_in, const int* in_sizes, int n_in,
                              void* d_out, int out_size, void* d_ws, size_t ws_size,
                              hipStream_t stream) {
  using namespace cfg;
  (void)in_sizes; (void)n_in; (void)out_size; (void)ws_size;

  const int*   x       = (const int*)d_in[0];
  const float* eps     = (const float*)d_in[1];
  const float* enc_emb = (const float*)d_in[2];
  const float* h0      = (const float*)d_in[3];
  const float* c0      = (const float*)d_in[4];
  const float* h0_2    = (const float*)d_in[5];
  const float* c0_2    = (const float*)d_in[6];
  const float* Wih1 = (const float*)d_in[7];  const float* Whh1 = (const float*)d_in[8];
  const float* bih1 = (const float*)d_in[9];  const float* bhh1 = (const float*)d_in[10];
  const float* Wih2 = (const float*)d_in[11]; const float* Whh2 = (const float*)d_in[12];
  const float* bih2 = (const float*)d_in[13]; const float* bhh2 = (const float*)d_in[14];
  const float* Wm1 = (const float*)d_in[15]; const float* bm1 = (const float*)d_in[16];
  const float* Wm2 = (const float*)d_in[17]; const float* bm2 = (const float*)d_in[18];
  const float* Ws1 = (const float*)d_in[19]; const float* bs1 = (const float*)d_in[20];
  const float* Ws2 = (const float*)d_in[21]; const float* bs2 = (const float*)d_in[22];
  const float* Wf1 = (const float*)d_in[23]; const float* bf1 = (const float*)d_in[24];
  const float* Wf2 = (const float*)d_in[25]; const float* bf2 = (const float*)d_in[26];
  const float* dec_emb = (const float*)d_in[27];
  const float* Wihd = (const float*)d_in[28]; const float* Whhd = (const float*)d_in[29];
  const float* bihd = (const float*)d_in[30]; const float* bhhd = (const float*)d_in[31];
  const float* Wfc  = (const float*)d_in[32]; const float* bfc  = (const float*)d_in[33];

  float* out = (float*)d_out;
  const size_t logits_n = (size_t)S * B * V;
  float* out_hc  = out + logits_n;
  float* out_rec = out_hc + (size_t)B * IO;
  float* out_z   = out_rec + (size_t)B * IO;

  // --- workspace carve-out ---
  uint8_t* ws = (uint8_t*)d_ws;
  size_t off = 0;
  auto carve = [&](size_t bytes) -> void* {
    off = (off + 255) & ~(size_t)255;
    void* p = ws + off;
    off += bytes;
    return p;
  };
  __bf16* Wcat1 = (__bf16*)carve((size_t)G * KC1 * 2);
  __bf16* Wcat2 = (__bf16*)carve((size_t)G * KC2 * 2);
  __bf16* Wcatd = (__bf16*)carve((size_t)G * KCD * 2);
  __bf16* Wm1b  = (__bf16*)carve((size_t)IO * IO * 2);
  __bf16* Wm2b  = (__bf16*)carve((size_t)L * IO * 2);
  __bf16* Ws1b  = (__bf16*)carve((size_t)IO * IO * 2);
  __bf16* Ws2b  = (__bf16*)carve((size_t)L * IO * 2);
  __bf16* Wf1b  = (__bf16*)carve((size_t)IO * L * 2);
  __bf16* Wf2b  = (__bf16*)carve((size_t)IO * IO * 2);
  __bf16* Wfcb  = (__bf16*)carve((size_t)V * H * 2);
  float* bcat1  = (float*)carve((size_t)G * 4);
  float* bcat2  = (float*)carve((size_t)G * 4);
  float* bcatd  = (float*)carve((size_t)G * 4);
  __bf16* Xenc  = (__bf16*)carve((size_t)S * B * E * 2);
  __bf16* Xdec  = (__bf16*)carve((size_t)S * B * E * 2);
  __bf16* A1    = (__bf16*)carve((size_t)B * KC1 * 2);
  __bf16* A2    = (__bf16*)carve((size_t)B * KC2 * 2);
  __bf16* Ad    = (__bf16*)carve((size_t)B * KCD * 2);
  float* g1     = (float*)carve((size_t)B * G * 4);
  float* g2     = (float*)carve((size_t)B * G * 4);
  float* gd     = (float*)carve((size_t)B * G * 4);
  float* h1a = (float*)carve((size_t)B * H * 4); float* h1b = (float*)carve((size_t)B * H * 4);
  float* c1a = (float*)carve((size_t)B * H * 4); float* c1b = (float*)carve((size_t)B * H * 4);
  float* h2a = (float*)carve((size_t)B * H * 4); float* h2b = (float*)carve((size_t)B * H * 4);
  float* c2a = (float*)carve((size_t)B * H * 4); float* c2b = (float*)carve((size_t)B * H * 4);
  float* hdbuf = (float*)carve((size_t)B * H * 4);
  float* cdbuf = (float*)carve((size_t)B * H * 4);
  __bf16* hcb  = (__bf16*)carve((size_t)B * IO * 2);
  __bf16* t1b  = (__bf16*)carve((size_t)B * IO * 2);
  __bf16* t3b  = (__bf16*)carve((size_t)B * IO * 2);
  float* meanf = (float*)carve((size_t)B * L * 4);
  float* stdf  = (float*)carve((size_t)B * L * 4);
  __bf16* zb   = (__bf16*)carve((size_t)B * L * 2);
  __bf16* hs   = (__bf16*)carve((size_t)S * B * H * 2);

  auto blk = [](long n) { return dim3((unsigned)((n + 255) / 256)); };

  // --- weight prep (bf16 pack/convert) ---
  k_pack2<<<blk((long)G * KC1), 256, 0, stream>>>(Wih1, Whh1, Wcat1, G, E, H);
  k_pack2<<<blk((long)G * KC2), 256, 0, stream>>>(Wih2, Whh2, Wcat2, G, H, H);
  k_pack2<<<blk((long)G * KCD), 256, 0, stream>>>(Wihd, Whhd, Wcatd, G, E, H);
  k_cvt_bf16<<<blk((long)IO * IO), 256, 0, stream>>>(Wm1, Wm1b, IO * IO);
  k_cvt_bf16<<<blk((long)L * IO), 256, 0, stream>>>(Wm2, Wm2b, L * IO);
  k_cvt_bf16<<<blk((long)IO * IO), 256, 0, stream>>>(Ws1, Ws1b, IO * IO);
  k_cvt_bf16<<<blk((long)L * IO), 256, 0, stream>>>(Ws2, Ws2b, L * IO);
  k_cvt_bf16<<<blk((long)IO * L), 256, 0, stream>>>(Wf1, Wf1b, IO * L);
  k_cvt_bf16<<<blk((long)IO * IO), 256, 0, stream>>>(Wf2, Wf2b, IO * IO);
  k_cvt_bf16<<<blk((long)V * H), 256, 0, stream>>>(Wfc, Wfcb, V * H);
  k_badd<<<blk(G), 256, 0, stream>>>(bih1, bhh1, bcat1, G);
  k_badd<<<blk(G), 256, 0, stream>>>(bih2, bhh2, bcat2, G);
  k_badd<<<blk(G), 256, 0, stream>>>(bihd, bhhd, bcatd, G);

  // --- embeddings ---
  k_embed_enc<<<blk((long)S * B * E), 256, 0, stream>>>(x, enc_emb, Xenc);
  k_embed_dec<<<blk((long)S * B * E), 256, 0, stream>>>(x, dec_emb, Xdec);

  // --- init encoder state + A matrices ---
  k_init<<<blk((long)B * H), 256, 0, stream>>>(h0, c0, h0_2, c0_2, h1a, c1a, h2a, c2a,
                                               A1, A2, Xenc);

  auto gemm = [&](const __bf16* Abuf, const __bf16* Wbuf, const float* bias, float* Cf,
                  __bf16* Cb, int M, int N, int K, int relu) {
    const int threads = (M >= 128) ? 256 : 128;       // 8 or 4 waves per block
    const int wpb = threads / 32;
    dim3 grid((unsigned)(N / 16), (unsigned)(M / (16 * wpb)));
    k_gemm<<<grid, threads, 0, stream>>>(Abuf, Wbuf, bias, Cf, Cb, M, N, K, relu);
  };

  // --- encoder: 128 sequential steps, ping-pong state buffers ---
  float *h1p = h1a, *c1p = c1a, *h1n = h1b, *c1n = c1b;
  float *h2p = h2a, *c2p = c2a, *h2n = h2b, *c2n = c2b;
  for (int t = 0; t < S; ++t) {
    gemm(A1, Wcat1, bcat1, g1, nullptr, B, G, KC1, 0);
    k_elem1<<<blk((long)B * H), 256, 0, stream>>>(g1, h1p, c1p, h1n, c1n, x, t, A2, A1, Xenc);
    gemm(A2, Wcat2, bcat2, g2, nullptr, B, G, KC2, 0);
    k_elem2<<<blk((long)B * H), 256, 0, stream>>>(g2, h2p, c2p, h2n, c2n, h1p, c1p, x, t, A2);
    float* tp;
    tp = h1p; h1p = h1n; h1n = tp;  tp = c1p; c1p = c1n; c1n = tp;
    tp = h2p; h2p = h2n; h2n = tp;  tp = c2p; c2p = c2n; c2n = tp;
  }

  // --- hc = [h2 | c2] ---
  k_hc<<<blk((long)B * IO), 256, 0, stream>>>(h2p, c2p, out_hc, hcb);

  // --- CVAE ---
  gemm(hcb, Wm1b, bm1, nullptr, t1b, B, IO, IO, 1);
  gemm(t1b, Wm2b, bm2, meanf, nullptr, B, L, IO, 0);
  gemm(hcb, Ws1b, bs1, nullptr, t1b, B, IO, IO, 1);
  gemm(t1b, Ws2b, bs2, stdf, nullptr, B, L, IO, 0);
  k_z<<<blk((long)B * L), 256, 0, stream>>>(meanf, stdf, eps, out_z, zb);
  gemm(zb, Wf1b, bf1, nullptr, t3b, B, IO, L, 1);
  gemm(t3b, Wf2b, bf2, out_rec, nullptr, B, IO, IO, 0);
  k_split<<<blk((long)B * H), 256, 0, stream>>>(out_rec, hdbuf, cdbuf, Ad, Xdec);

  // --- decoder: 128 sequential steps ---
  for (int t = 0; t < S; ++t) {
    gemm(Ad, Wcatd, bcatd, gd, nullptr, B, G, KCD, 0);
    k_elemd<<<blk((long)B * H), 256, 0, stream>>>(gd, hdbuf, cdbuf, t, Ad, Xdec, hs);
  }

  // --- logits = hs @ Wfc^T + bfc  (8192 x 16000 x 1024) ---
  gemm(hs, Wfcb, bfc, out, nullptr, S * B, V, H, 0);
}
